// ContentBasedAttention_50551765074232
// MI455X (gfx1250) — matrix operand order repaired and checked by reading, not compiled, not run on hardware
//
#include <hip/hip_runtime.h>

typedef __attribute__((ext_vector_type(16))) __bf16 v16bf;
typedef __attribute__((ext_vector_type(8)))  __bf16 v8bf;
typedef __attribute__((ext_vector_type(4)))  __bf16 v4bf;
typedef __attribute__((ext_vector_type(8)))  float  v8f;
typedef __attribute__((ext_vector_type(4)))  float  v4f;
typedef __attribute__((ext_vector_type(4)))  unsigned int u32x4;
typedef __attribute__((ext_vector_type(8)))  int    i32x8;
typedef __attribute__((ext_vector_type(4)))  int    i32x4;

#define H_DIM 512
#define N_BATCH 32
#define TI 4096
#define WAVES 4
#define THREADS 128                 // 4 waves * 32 lanes (wave32)
#define ROWS_PER_WAVE 32            // 2 row-tiles of 16 per wave
#define ROWS_PER_BLOCK (WAVES*ROWS_PER_WAVE)  // 128
#define COLS_PER_ITER 32            // 2 col-tiles of 16 per outer iteration
#define OUTER_ITERS (H_DIM/COLS_PER_ITER)     // 16
#define K_STEPS (H_DIM/32)          // 16 WMMA k-steps (K=32 each)

#define LDSA_BYTES (ROWS_PER_BLOCK * H_DIM * 2)          // 128 KB
#define LDSB_BYTES (COLS_PER_ITER * H_DIM * 2)           //  32 KB per buffer
#define SMEM_BYTES (LDSA_BYTES + 2 * LDSB_BYTES)         // 192 KB

#if defined(__HIP_DEVICE_COMPILE__) && defined(__gfx1250__) && \
    __has_builtin(__builtin_amdgcn_tensor_load_to_lds) && \
    __has_builtin(__builtin_amdgcn_s_wait_tensorcnt)
#define USE_TDM 1
#endif

// ---------------------------------------------------------------------------
// Kernel 0a: convert Wv (f32, row-major [o][h]) to bf16 in workspace.
// ---------------------------------------------------------------------------
__global__ __launch_bounds__(256)
void cvt_wv_kernel(const float* __restrict__ Wv, __bf16* __restrict__ WvBf) {
    int idx = blockIdx.x * 256 + threadIdx.x;   // H*H = 262144 total
    WvBf[idx] = (__bf16)Wv[idx];
}

// ---------------------------------------------------------------------------
// Kernel 0b: Ws[n][o] = dot(dec[n,:], Ww[o,:]) + bw[o]   (N*H = 16384 outputs)
// ---------------------------------------------------------------------------
__global__ __launch_bounds__(256)
void ws_kernel(const float* __restrict__ dec, const float* __restrict__ Ww,
               const float* __restrict__ bw, float* __restrict__ WsBuf) {
    int idx = blockIdx.x * 256 + threadIdx.x;
    int n = idx >> 9;
    int o = idx & (H_DIM - 1);
    const float* d = dec + (size_t)n * H_DIM;
    const float* w = Ww + (size_t)o * H_DIM;
    float s = bw[o];
    #pragma unroll 4
    for (int h = 0; h < H_DIM; ++h) s += d[h] * w[h];
    WsBuf[idx] = s;
}

// ---------------------------------------------------------------------------
// TDM: load a 32-row x 512-col bf16 tile of Wv into LDS.
// D# per CDNA5 ISA 8.3/8.4: group0 = {count=1, lds_addr, global_addr, type=2},
// group1 = {data_size=2B, tensor 512x512, tile 512x32, dim0_stride=512}.
// Groups 2/3 (3D-5D only) and the trailing group are zero (2D tensor).
// Issued by one wave (EXEC is ignored for tensor ops); tracked by TENSORcnt.
// ---------------------------------------------------------------------------
#ifdef USE_TDM
__device__ __forceinline__ void tdm_load_b_tile(unsigned lds_addr, const __bf16* gsrc) {
    unsigned long long ga = (unsigned long long)(uintptr_t)gsrc;
    u32x4 g0;
    g0[0] = 1u;                                        // count=1 (valid), user mode
    g0[1] = lds_addr;                                  // LDS byte address
    g0[2] = (unsigned)(ga & 0xffffffffull);            // global_addr[31:0]
    g0[3] = (unsigned)((ga >> 32) & 0x1ffffffull)      // global_addr[56:32]
          | (2u << 30);                                // type=2 ("image")
    i32x8 g1;
    g1[0] = (int)(1u << 16);          // workgroup_mask=0, data_size=1 (2 bytes)
    g1[1] = (int)(512u << 16);        // tensor_dim0[15:0]=512 (bits 63:48)
    g1[2] = (int)(512u << 16);        // tensor_dim0[31:16]=0 | tensor_dim1[15:0]=512
    g1[3] = (int)(512u << 16);        // tensor_dim1[31:16]=0 | tile_dim0=512
    g1[4] = 32;                       // tile_dim1=32, tile_dim2=0
    g1[5] = 512;                      // tensor_dim0_stride[31:0]=512 (elements)
    g1[6] = 0;                        // stride0[47:32]=0, dim1_stride[15:0]=0 (2D)
    g1[7] = 0;
    i32x4 z4 = {0, 0, 0, 0};
    i32x8 z8 = {0, 0, 0, 0, 0, 0, 0, 0};
    __builtin_amdgcn_tensor_load_to_lds(g0, g1, z4, z4, z8, 0);
}
#endif

// ---------------------------------------------------------------------------
// Kernel 1: fused e[n,t] = where(mask, -1e-5, wT . tanh(enc@Wv^T + Ws[n]))
// 2x2 register tiling: each wave computes 32 rows x 2 col-tiles per iteration
// (4 WMMA accumulators, every A/B fragment reused twice -> halves DS traffic).
// B tiles double-buffered in LDS, prefetched by the Tensor Data Mover.
// ---------------------------------------------------------------------------
__global__ __launch_bounds__(THREADS)
void attn_e_kernel(const float* __restrict__ enc, const __bf16* __restrict__ WvBf,
                   const float* __restrict__ WsBuf, const float* __restrict__ wT,
                   const unsigned char* __restrict__ mask, float* __restrict__ eBuf) {
    extern __shared__ char smem[];
    __bf16* ldsA  = (__bf16*)smem;                       // 128 rows * 512 = 128KB
    __bf16* ldsB0 = (__bf16*)(smem + LDSA_BYTES);        // 32 cols * 512 = 32KB
    __bf16* ldsB1 = (__bf16*)(smem + LDSA_BYTES + LDSB_BYTES);

    const int tid  = threadIdx.x;
    const int lane = tid & 31;
    const int wave = tid >> 5;
    const int hl   = lane >> 4;     // 0: lanes 0-15, 1: lanes 16-31
    const int mrow = lane & 15;

    const int row0     = blockIdx.x * ROWS_PER_BLOCK;  // global row = n*TI + t
    const int n        = row0 / TI;                    // TI % ROWS_PER_BLOCK == 0
    const int waveRow0 = row0 + wave * ROWS_PER_WAVE;

#ifdef USE_TDM
    const unsigned ldsB0_addr = (unsigned)(uintptr_t)(void*)ldsB0;
    const unsigned ldsB1_addr = (unsigned)(uintptr_t)(void*)ldsB1;
    // Prefetch B tile 0 into buffer 0 (wave 0 only, overlaps with A staging).
    if (wave == 0) tdm_load_b_tile(ldsB0_addr, WvBf);
#else
    {   // Fallback: synchronous cooperative copy of B tile 0.
        const unsigned int* src = (const unsigned int*)WvBf;
        unsigned int* dst = (unsigned int*)ldsB0;
        for (int i = tid; i < COLS_PER_ITER * H_DIM / 2; i += THREADS) dst[i] = src[i];
    }
#endif

    // --- Stage this wave's 32x512 A-tile (f32 -> bf16) into its LDS region ---
    for (int r = 0; r < ROWS_PER_WAVE; ++r) {
        const float* src = enc + (size_t)(waveRow0 + r) * H_DIM;
        __bf16* dst = ldsA + (size_t)(wave * ROWS_PER_WAVE + r) * H_DIM;
        #pragma unroll
        for (int it = 0; it < 4; ++it) {
            int c = lane * 4 + it * 128;
            v4f v = *(const v4f*)(src + c);
            v4bf pv = { (__bf16)v.x, (__bf16)v.y, (__bf16)v.z, (__bf16)v.w };
            *(v4bf*)(dst + c) = pv;
        }
    }

    float epart0[8], epart1[8];
    #pragma unroll
    for (int j = 0; j < 8; ++j) { epart0[j] = 0.0f; epart1[j] = 0.0f; }

    // Per-lane fragment bases (contiguous k order used by this code, same for A and B):
    //  A (16-bit 16x32): lane<16 holds M=lane, K = k0+0..15;
    //                    lane>=16 holds M=lane-16, K = k0+16..31.
    //  B (16-bit 32x16): lanes 0-15 hold K=k0+0..15 for N=lane;
    //                    lanes 16-31 hold K=k0+16..31 for N=lane-16.
    const __bf16* aBase0 = ldsA + (size_t)(wave * ROWS_PER_WAVE + mrow) * H_DIM + hl * 16;
    const __bf16* aBase1 = aBase0 + 16 * H_DIM;

#ifdef USE_TDM
    if (wave == 0) __builtin_amdgcn_s_wait_tensorcnt(0);
#endif
    __syncthreads();

    for (int ot = 0; ot < OUTER_ITERS; ++ot) {
        __bf16* bufCur = (ot & 1) ? ldsB1 : ldsB0;
        // Prefetch next 32-col B tile into the other buffer while computing.
        if (ot + 1 < OUTER_ITERS) {
            const __bf16* gsrc = WvBf + (size_t)(ot + 1) * COLS_PER_ITER * H_DIM;
#ifdef USE_TDM
            if (wave == 0)
                tdm_load_b_tile((ot & 1) ? ldsB0_addr : ldsB1_addr, gsrc);
#else
            unsigned int* dst = (unsigned int*)((ot & 1) ? ldsB0 : ldsB1);
            const unsigned int* src = (const unsigned int*)gsrc;
            for (int i = tid; i < COLS_PER_ITER * H_DIM / 2; i += THREADS) dst[i] = src[i];
#endif
        }

        const __bf16* bBase0 = bufCur + (size_t)mrow * H_DIM + hl * 16;
        const __bf16* bBase1 = bBase0 + 16 * H_DIM;

        v8f acc00 = {}, acc01 = {}, acc10 = {}, acc11 = {};
        #pragma unroll
        for (int ks = 0; ks < K_STEPS; ++ks) {
            const int k0 = ks * 32;
            v8bf a0lo = *(const v8bf*)(aBase0 + k0);
            v8bf a0hi = *(const v8bf*)(aBase0 + k0 + 8);
            v16bf af0 = __builtin_shufflevector(a0lo, a0hi,
                0, 1, 2, 3, 4, 5, 6, 7, 8, 9, 10, 11, 12, 13, 14, 15);
            v8bf a1lo = *(const v8bf*)(aBase1 + k0);
            v8bf a1hi = *(const v8bf*)(aBase1 + k0 + 8);
            v16bf af1 = __builtin_shufflevector(a1lo, a1hi,
                0, 1, 2, 3, 4, 5, 6, 7, 8, 9, 10, 11, 12, 13, 14, 15);
            v16bf bf0 = *(const v16bf*)(bBase0 + k0);
            v16bf bf1 = *(const v16bf*)(bBase1 + k0);
            acc00 = __builtin_amdgcn_wmma_f32_16x16x32_bf16(false, af0, false, bf0, (short)0, acc00, false, false);
            acc01 = __builtin_amdgcn_wmma_f32_16x16x32_bf16(false, af0, false, bf1, (short)0, acc01, false, false);
            acc10 = __builtin_amdgcn_wmma_f32_16x16x32_bf16(false, af1, false, bf0, (short)0, acc10, false, false);
            acc11 = __builtin_amdgcn_wmma_f32_16x16x32_bf16(false, af1, false, bf1, (short)0, acc11, false, false);
        }

        // Epilogue: acc(r,c)[j] = Vh[row = waveRow0 + 16*r + 8*hl + j][o = ot*32 + 16*c + mrow]
        const int o0 = ot * COLS_PER_ITER + mrow;
        const int o1 = o0 + 16;
        const float wsv0 = WsBuf[n * H_DIM + o0];
        const float wtv0 = wT[o0];
        const float wsv1 = WsBuf[n * H_DIM + o1];
        const float wtv1 = wT[o1];
        #pragma unroll
        for (int j = 0; j < 8; ++j) {
            epart0[j] += tanhf(acc00[j] + wsv0) * wtv0 + tanhf(acc01[j] + wsv1) * wtv1;
            epart1[j] += tanhf(acc10[j] + wsv0) * wtv0 + tanhf(acc11[j] + wsv1) * wtv1;
        }

#ifdef USE_TDM
        if (wave == 0) __builtin_amdgcn_s_wait_tensorcnt(0);
#endif
        __syncthreads();
    }

    // Reduce across the 16 lanes of each half (sums over all 512 o-columns).
    #pragma unroll
    for (int j = 0; j < 8; ++j) {
        float v0 = epart0[j];
        v0 += __shfl_xor(v0, 1, 32);
        v0 += __shfl_xor(v0, 2, 32);
        v0 += __shfl_xor(v0, 4, 32);
        v0 += __shfl_xor(v0, 8, 32);
        epart0[j] = v0;
        float v1 = epart1[j];
        v1 += __shfl_xor(v1, 1, 32);
        v1 += __shfl_xor(v1, 2, 32);
        v1 += __shfl_xor(v1, 4, 32);
        v1 += __shfl_xor(v1, 8, 32);
        epart1[j] = v1;
    }
    if (mrow == 0) {
        #pragma unroll
        for (int j = 0; j < 8; ++j) {
            int g = waveRow0 + hl * 8 + j;      // row tile 0
            int t = g & (TI - 1);
            eBuf[g] = mask[(size_t)n * TI + t] ? -1e-5f : epart0[j];
            g += 16;                            // row tile 1
            t = g & (TI - 1);
            eBuf[g] = mask[(size_t)n * TI + t] ? -1e-5f : epart1[j];
        }
    }
}

// ---------------------------------------------------------------------------
// Kernel 2: row softmax over Ti=4096, one block per n, write a to d_out.
// ---------------------------------------------------------------------------
__global__ __launch_bounds__(256)
void softmax_kernel(const float* __restrict__ eBuf, float* __restrict__ aOut) {
    __shared__ float red[256];
    const int n = blockIdx.x, tid = threadIdx.x;
    const float* e = eBuf + (size_t)n * TI;

    float m = -1e30f;
    for (int t = tid; t < TI; t += 256) m = fmaxf(m, e[t]);
    red[tid] = m; __syncthreads();
    for (int s = 128; s > 0; s >>= 1) {
        if (tid < s) red[tid] = fmaxf(red[tid], red[tid + s]);
        __syncthreads();
    }
    const float gmax = red[0]; __syncthreads();

    float sum = 0.0f;
    for (int t = tid; t < TI; t += 256) sum += expf(e[t] - gmax);
    red[tid] = sum; __syncthreads();
    for (int s = 128; s > 0; s >>= 1) {
        if (tid < s) red[tid] += red[tid + s];
        __syncthreads();
    }
    const float inv = 1.0f / red[0];

    float* a = aOut + (size_t)n * TI;
    for (int t = tid; t < TI; t += 256) a[t] = expf(e[t] - gmax) * inv;
}

// ---------------------------------------------------------------------------
// Kernel 3: partial context: cPart[n][chunk][h] = sum_{t in chunk} a[n,t]*enc[n,t,h]
// Grid (16 chunks, 32 n), 256 threads (each handles 2 h's). Deterministic.
// ---------------------------------------------------------------------------
__global__ __launch_bounds__(256)
void cpart_kernel(const float* __restrict__ enc, const float* __restrict__ aOut,
                  float* __restrict__ cPart) {
    __shared__ float aS[256];
    const int n = blockIdx.y, chunk = blockIdx.x, tid = threadIdx.x;
    const int t0 = chunk * 256;
    aS[tid] = aOut[(size_t)n * TI + t0 + tid];
    __syncthreads();

    const float* base = enc + ((size_t)n * TI + t0) * H_DIM;
    float acc0 = 0.0f, acc1 = 0.0f;
    for (int tt = 0; tt < 256; ++tt) {
        const float av = aS[tt];
        acc0 += av * base[(size_t)tt * H_DIM + tid];
        acc1 += av * base[(size_t)tt * H_DIM + tid + 256];
    }
    float* dst = cPart + (size_t)(n * 16 + chunk) * H_DIM;
    dst[tid] = acc0;
    dst[tid + 256] = acc1;
}

// ---------------------------------------------------------------------------
// Kernel 4: c[n][h] = sum over 16 chunks (fixed order).
// ---------------------------------------------------------------------------
__global__ __launch_bounds__(256)
void creduce_kernel(const float* __restrict__ cPart, float* __restrict__ cOut) {
    const int idx = blockIdx.x * 256 + threadIdx.x;   // N*H = 16384
    const int n = idx >> 9, h = idx & (H_DIM - 1);
    float s = 0.0f;
    #pragma unroll
    for (int ch = 0; ch < 16; ++ch) s += cPart[(size_t)(n * 16 + ch) * H_DIM + h];
    cOut[idx] = s;
}

// ---------------------------------------------------------------------------
extern "C" void kernel_launch(void* const* d_in, const int* in_sizes, int n_in,
                              void* d_out, int out_size, void* d_ws, size_t ws_size,
                              hipStream_t stream) {
    const float*         dec  = (const float*)d_in[0];          // [N,1,H]
    const float*         enc  = (const float*)d_in[1];          // [N,Ti,H]
    const unsigned char* mask = (const unsigned char*)d_in[2];  // [N,Ti] bool
    const float*         Wv   = (const float*)d_in[3];          // [H,H]
    const float*         Ww   = (const float*)d_in[4];          // [H,H]
    const float*         bw   = (const float*)d_in[5];          // [H]
    const float*         wT   = (const float*)d_in[6];          // [H]

    float* out  = (float*)d_out;
    float* cOut = out;                       // c: N*H floats first
    float* aOut = out + N_BATCH * H_DIM;     // a: N*Ti floats

    // Workspace layout
    char* ws = (char*)d_ws;
    __bf16* WvBf  = (__bf16*)ws;                                   // 512 KB
    float*  WsBuf = (float*)(ws + 524288);                         //  64 KB
    float*  eBuf  = (float*)(ws + 524288 + 65536);                 // 512 KB
    float*  cPart = (float*)(ws + 524288 + 65536 + 524288);        //   1 MB

    cvt_wv_kernel<<<(H_DIM * H_DIM) / 256, 256, 0, stream>>>(Wv, WvBf);
    ws_kernel<<<(N_BATCH * H_DIM) / 256, 256, 0, stream>>>(dec, Ww, bw, WsBuf);

    attn_e_kernel<<<(N_BATCH * TI) / ROWS_PER_BLOCK, THREADS, SMEM_BYTES, stream>>>(
        enc, WvBf, WsBuf, wT, mask, eBuf);

    softmax_kernel<<<N_BATCH, 256, 0, stream>>>(eBuf, aOut);

    dim3 gPart(16, N_BATCH);
    cpart_kernel<<<gPart, 256, 0, stream>>>(enc, aOut, cPart);
    creduce_kernel<<<(N_BATCH * H_DIM) / 256, 256, 0, stream>>>(cPart, cOut);
}
